// XEncoder_3882650436757
// MI455X (gfx1250) — matrix operand order repaired and checked by reading, not compiled, not run on hardware
//
#include <hip/hip_runtime.h>
#include <math.h>

#define TT   2048
#define BB   8
#define DD   1024
#define HIDD 1024
#define OUTD 512
#define NH   4
#define DK   256

typedef __bf16 bf16_t;
typedef __attribute__((ext_vector_type(16))) __bf16 v16bf;
typedef __attribute__((ext_vector_type(8)))  float  v8f;
typedef __attribute__((ext_vector_type(4)))  unsigned u32x4;
typedef __attribute__((ext_vector_type(8)))  int      i32x8;
typedef __attribute__((ext_vector_type(4)))  int      i32x4;

#if defined(__has_builtin)
#if __has_builtin(__builtin_amdgcn_tensor_load_to_lds) && __has_builtin(__builtin_amdgcn_s_wait_tensorcnt)
#define USE_TDM 1
#endif
#endif
#ifndef USE_TDM
#define USE_TDM 0
#endif

union FragBF { v16bf v; unsigned u[8]; };

__device__ __forceinline__ unsigned pack2_bf16(float x, float y) {
  unsigned ux = __builtin_bit_cast(unsigned, x);
  unsigned uy = __builtin_bit_cast(unsigned, y);
  ux += 0x7FFFu + ((ux >> 16) & 1u);   // RNE
  uy += 0x7FFFu + ((uy >> 16) & 1u);
  return (ux >> 16) | (uy & 0xFFFF0000u);
}

__device__ __forceinline__ float gelu_exact(float x) {
  return 0.5f * x * (1.0f + erff(x * 0.70710678118654752f));
}

#if USE_TDM
// 2D TDM tile load: bf16 tile (tileRows x tileCols elements) from a row-major
// matrix with rowStride elements/row into contiguous LDS rows.
// D# layout per CDNA5 ISA.
// This toolchain exposes the 6-arg builtin:
//   (uint32x4 g0, int32x8 g1, int32x4 g2, int32x4 g3, int32x8 pad, i32 cpol)
__device__ __forceinline__ void tdm_load_2d(const bf16_t* g, unsigned ldsAddr,
                                            unsigned tileCols, unsigned tileRows,
                                            unsigned strideElems) {
  unsigned long long ga = (unsigned long long)(size_t)g;
  u32x4 g0;
  g0[0] = 1u;                                   // count=1, user descriptor
  g0[1] = ldsAddr;                              // LDS byte address
  g0[2] = (unsigned)ga;                         // global_addr[31:0]
  g0[3] = ((unsigned)(ga >> 32) & 0x01FFFFFFu)  // global_addr[56:32]
          | 0x80000000u;                        // type=2 ("image")
  i32x8 g1;
  g1[0] = 0x00010000;                           // wg_mask=0, data_size=1 (2B)
  g1[1] = 0;                                    // atomic_addr=0, tensor_dim0[15:0]=0
  g1[2] = 0x00004000;                           // tensor_dim0=2^30, tensor_dim1[15:0]=0
  g1[3] = (int)(0x00004000u | (tileCols << 16));// tensor_dim1=2^30, tile_dim0
  g1[4] = (int)tileRows;                        // tile_dim1; tile_dim2=0
  g1[5] = (int)strideElems;                     // tensor_dim0_stride[31:0]
  g1[6] = 0;                                    // stride0 hi / stride1 lo
  g1[7] = 0;
  i32x4 z4 = {0, 0, 0, 0};
  i32x8 z8 = {0, 0, 0, 0, 0, 0, 0, 0};
  __builtin_amdgcn_tensor_load_to_lds(g0, g1, z4, z4, z8, 0);
}
#endif

__device__ __forceinline__ void mma_step(const unsigned* __restrict__ Au,
                                         const unsigned* __restrict__ Bu,
                                         int wm, int wn, int half, int l16,
                                         v8f acc[2][2]) {
  #pragma unroll
  for (int ks = 0; ks < 2; ++ks) {                 // two K=32 substeps of the K=64 tile
    FragBF a[2], b[2];
    #pragma unroll
    for (int im = 0; im < 2; ++im) {
      const int row = wm * 32 + im * 16 + l16;
      #pragma unroll
      for (int i = 0; i < 8; ++i) {
        const int pair = ks * 16 + ((i < 4) ? (half * 4 + i) : (8 + half * 4 + (i - 4)));
        a[im].u[i] = Au[row * 32 + pair];
      }
    }
    #pragma unroll
    for (int in = 0; in < 2; ++in) {
      const int col = wn * 32 + in * 16 + l16;
      #pragma unroll
      for (int i = 0; i < 8; ++i)
        b[in].u[i] = Bu[col * 32 + ks * 16 + ((i < 4) ? (half * 4 + i) : (8 + half * 4 + (i - 4)))];
    }
    #pragma unroll
    for (int im = 0; im < 2; ++im)
      #pragma unroll
      for (int in = 0; in < 2; ++in)
        acc[im][in] = __builtin_amdgcn_wmma_f32_16x16x32_bf16(
            false, a[im].v, false, b[in].v, (short)0, acc[im][in], false, false);
  }
}

// C[M,N] = A[M,K] @ Wt[N,K]^T   (A, Wt bf16 row-major; fp32 accumulate)
// epilogue: + bias[n]; optional +Res[m*N+n]; optional exact GELU;
// ROWMODE: 0 none, 1 f32 row-major, 2 bf16 row-major; optional [B,N,T] f32 store.
template<int ROWMODE, bool DOGELU, bool DORES, bool STORE_BOT>
__global__ __launch_bounds__(256)
void gemm_bf16(const bf16_t* __restrict__ A, const bf16_t* __restrict__ Wt,
               const float* __restrict__ bias, const float* __restrict__ Res,
               float* __restrict__ OutRowF, bf16_t* __restrict__ OutRowH,
               float* __restrict__ OutBot, int M, int N, int K) {
  constexpr int TA = 128 * 64;                 // A tile elements (16 KB)
  constexpr int TB = 64 * 64;                  // B tile elements (8 KB)
  __shared__ __align__(16) bf16_t smem[(TA + TB) * 2];   // double buffered: 48 KB

  const int tid  = threadIdx.x;
  const int lane = tid & 31;
  const int wave = tid >> 5;
  const int half = lane >> 4;
  const int l16  = lane & 15;
  const int wm   = wave & 3;
  const int wn   = wave >> 2;
  const int m0   = blockIdx.y * 128;
  const int n0   = blockIdx.x * 64;
  const int nk   = K >> 6;

  v8f acc[2][2];
  #pragma unroll
  for (int i = 0; i < 2; ++i)
    #pragma unroll
    for (int j = 0; j < 2; ++j) acc[i][j] = (v8f){};

#if USE_TDM
  const unsigned ldsBase = (unsigned)(unsigned long long)(size_t)&smem[0];
  if (wave == 0) {   // wave-uniform branch; TDM is a wave-level DMA op
    tdm_load_2d(A  + (size_t)m0 * K, ldsBase,               64, 128, (unsigned)K);
    tdm_load_2d(Wt + (size_t)n0 * K, ldsBase + TA * 2,      64,  64, (unsigned)K);
  }
  for (int i = 0; i < nk; ++i) {
    const int cur = i & 1;
    if (wave == 0) {
      if (i + 1 < nk) {      // prefetch next K-panel into the other buffer
        const unsigned off = (unsigned)(((i + 1) & 1) * (TA + TB) * 2);
        tdm_load_2d(A  + (size_t)m0 * K + (size_t)(i + 1) * 64, ldsBase + off,          64, 128, (unsigned)K);
        tdm_load_2d(Wt + (size_t)n0 * K + (size_t)(i + 1) * 64, ldsBase + off + TA * 2, 64,  64, (unsigned)K);
        __builtin_amdgcn_s_wait_tensorcnt((short)2);   // oldest 2 (current buffer) done
      } else {
        __builtin_amdgcn_s_wait_tensorcnt((short)0);
      }
    }
    __syncthreads();         // publish DMA'd tiles to all waves
    const unsigned* Au = (const unsigned*)(smem + cur * (TA + TB));
    const unsigned* Bu = (const unsigned*)(smem + cur * (TA + TB) + TA);
    mma_step(Au, Bu, wm, wn, half, l16, acc);
    __syncthreads();         // all waves done with 'cur' before it is overwritten
  }
#else
  // Fallback: synchronous cooperative b128 copies (single buffer).
  for (int i = 0; i < nk; ++i) {
    const int k0 = i * 64;
    __syncthreads();
    #pragma unroll
    for (int it = 0; it < 4; ++it) {             // A tile: 1024 x 16B
      const int idx = tid + it * 256;
      const int row = idx >> 3, q = idx & 7;
      const uint4 d = *reinterpret_cast<const uint4*>(A + (size_t)(m0 + row) * K + k0 + q * 8);
      *reinterpret_cast<uint4*>(smem + row * 64 + q * 8) = d;
    }
    #pragma unroll
    for (int it = 0; it < 2; ++it) {             // B tile: 512 x 16B
      const int idx = tid + it * 256;
      const int row = idx >> 3, q = idx & 7;
      const uint4 d = *reinterpret_cast<const uint4*>(Wt + (size_t)(n0 + row) * K + k0 + q * 8);
      *reinterpret_cast<uint4*>(smem + TA + row * 64 + q * 8) = d;
    }
    __syncthreads();
    mma_step((const unsigned*)smem, (const unsigned*)(smem + TA), wm, wn, half, l16, acc);
  }
#endif

  // ---- epilogue
  #pragma unroll
  for (int im = 0; im < 2; ++im) {
    #pragma unroll
    for (int in = 0; in < 2; ++in) {
      const int ng = n0 + wn * 32 + in * 16 + l16;
      const float bv = bias[ng];
      #pragma unroll
      for (int r = 0; r < 8; ++r) {
        const int mg = m0 + wm * 32 + im * 16 + half * 8 + r;
        float val = acc[im][in][r] + bv;
        if (DORES) val += Res[(size_t)mg * N + ng];
        if (DOGELU) val = gelu_exact(val);
        if (ROWMODE == 1) OutRowF[(size_t)mg * N + ng] = val;
        if (ROWMODE == 2) OutRowH[(size_t)mg * N + ng] = (bf16_t)val;
        if (STORE_BOT) {
          const int bb = mg >> 11;               // / T
          const int t  = mg & (TT - 1);
          OutBot[((size_t)bb * N + ng) * TT + t] = val;
        }
      }
    }
  }
}

// Windowed multi-head attention (band |i-j| <= 4), bf16 in/out, fp32 math.
__global__ __launch_bounds__(256)
void attn_window(const bf16_t* __restrict__ q, const bf16_t* __restrict__ kmat,
                 const bf16_t* __restrict__ vmat, bf16_t* __restrict__ o,
                 const float* __restrict__ adj_w, const float* __restrict__ adj_b) {
  const int lane = threadIdx.x & 31;
  const int wid  = (int)((blockIdx.x * blockDim.x + threadIdx.x) >> 5);
  const int t  = wid & (TT - 1);
  const int bh = wid >> 11;
  const int h  = bh & (NH - 1);
  const int b  = bh >> 2;
  const float gamma = adj_w[0];
  const float beta  = adj_b[0];
  const size_t rowq = (size_t)(b * TT + t) * HIDD + h * DK;

  float qr[8];
  #pragma unroll
  for (int i = 0; i < 8; ++i) qr[i] = (float)q[rowq + lane + i * 32];

  float s[9];
  float mx = -3.0e38f;
  #pragma unroll
  for (int jj = 0; jj < 9; ++jj) {
    const int j = t + jj - 4;                    // wave-uniform
    if (j >= 0 && j < TT) {
      const size_t rk = (size_t)(b * TT + j) * HIDD + h * DK;
      float p = 0.0f;
      #pragma unroll
      for (int i = 0; i < 8; ++i) p += qr[i] * (float)kmat[rk + lane + i * 32];
      #pragma unroll
      for (int off = 16; off > 0; off >>= 1) p += __shfl_xor(p, off, 32);
      const float d   = fabsf((float)(jj - 4));
      const float adj = expf(-fabsf(gamma * d * d - beta));
      s[jj] = p * 0.0625f + adj;                 // 1/sqrt(256)
    } else {
      s[jj] = -3.0e38f;
    }
    mx = fmaxf(mx, s[jj]);
  }

  float e[9], Z = 0.0f;
  #pragma unroll
  for (int jj = 0; jj < 9; ++jj) {
    e[jj] = (s[jj] < -1.0e37f) ? 0.0f : expf(s[jj] - mx);
    Z += e[jj];
  }
  const float rz = 1.0f / Z;

  float accv[8];
  #pragma unroll
  for (int i = 0; i < 8; ++i) accv[i] = 0.0f;
  #pragma unroll
  for (int jj = 0; jj < 9; ++jj) {
    if (e[jj] != 0.0f) {
      const int j = t + jj - 4;
      const size_t rv = (size_t)(b * TT + j) * HIDD + h * DK;
      const float w = e[jj] * rz;
      #pragma unroll
      for (int i = 0; i < 8; ++i) accv[i] += w * (float)vmat[rv + lane + i * 32];
    }
  }
  #pragma unroll
  for (int i = 0; i < 8; ++i) o[rowq + lane + i * 32] = (bf16_t)accv[i];
}

// LayerNorm over D=1024 (fp32 in, bf16 out), one 256-thread block per row.
__global__ __launch_bounds__(256)
void layernorm_k(const float* __restrict__ yin, const float* __restrict__ g,
                 const float* __restrict__ bta, bf16_t* __restrict__ yout) {
  __shared__ float red[8], red2[8];
  const int row = blockIdx.x;
  const int tid = threadIdx.x;
  const float* p = yin + (size_t)row * DD;

  float v[4];
  float sum = 0.0f, sq = 0.0f;
  #pragma unroll
  for (int i = 0; i < 4; ++i) {
    v[i] = p[tid + i * 256];
    sum += v[i];
    sq  += v[i] * v[i];
  }
  #pragma unroll
  for (int off = 16; off > 0; off >>= 1) {
    sum += __shfl_xor(sum, off, 32);
    sq  += __shfl_xor(sq,  off, 32);
  }
  if ((tid & 31) == 0) { red[tid >> 5] = sum; red2[tid >> 5] = sq; }
  __syncthreads();
  sum = 0.0f; sq = 0.0f;
  #pragma unroll
  for (int i = 0; i < 8; ++i) { sum += red[i]; sq += red2[i]; }
  const float mu   = sum * (1.0f / DD);
  const float var  = sq * (1.0f / DD) - mu * mu;
  const float rstd = rsqrtf(var + 1e-5f);

  bf16_t* out = yout + (size_t)row * DD;
  #pragma unroll
  for (int i = 0; i < 4; ++i) {
    const int idx = tid + i * 256;
    out[idx] = (bf16_t)((v[i] - mu) * rstd * g[idx] + bta[idx]);
  }
}

// Bulk fp32 -> bf16 convert (4 elems / thread, exact grid).
__global__ __launch_bounds__(256)
void f32_to_bf16(const float* __restrict__ in, bf16_t* __restrict__ out) {
  const size_t i = (size_t)blockIdx.x * 256 + threadIdx.x;
  const float4 f = reinterpret_cast<const float4*>(in)[i];
  uint2 u;
  u.x = pack2_bf16(f.x, f.y);
  u.y = pack2_bf16(f.z, f.w);
  reinterpret_cast<uint2*>(out)[i] = u;
}

// Transpose-convert: Wt[n][k] = bf16(W[k][n]); 32x32 LDS tiles.
__global__ __launch_bounds__(256)
void transpose_to_bf16(const float* __restrict__ W, bf16_t* __restrict__ Wt,
                       int K, int N) {
  __shared__ float t[32][33];
  const int k0 = blockIdx.y * 32, n0 = blockIdx.x * 32;
  const int tid = threadIdx.x;
  const int r = tid >> 5, c = tid & 31;
  #pragma unroll
  for (int it = 0; it < 4; ++it)
    t[r + it * 8][c] = W[(size_t)(k0 + r + it * 8) * N + n0 + c];
  __syncthreads();
  #pragma unroll
  for (int it = 0; it < 4; ++it)
    Wt[(size_t)(n0 + r + it * 8) * K + k0 + c] = (bf16_t)t[c][r + it * 8];
}

extern "C" void kernel_launch(void* const* d_in, const int* in_sizes, int n_in,
                              void* d_out, int out_size, void* d_ws, size_t ws_size,
                              hipStream_t stream) {
  const float* x   = (const float*)d_in[0];
  // d_in[1] = seq_len (unused by the reference computation)
  const float* Wq  = (const float*)d_in[2];
  const float* bq  = (const float*)d_in[3];
  const float* Wk  = (const float*)d_in[4];
  const float* bk  = (const float*)d_in[5];
  const float* Wv  = (const float*)d_in[6];
  const float* bv  = (const float*)d_in[7];
  const float* Wo  = (const float*)d_in[8];
  const float* bo  = (const float*)d_in[9];
  const float* lng = (const float*)d_in[10];
  const float* lnb = (const float*)d_in[11];
  const float* W1  = (const float*)d_in[12];
  const float* b1  = (const float*)d_in[13];
  const float* W2  = (const float*)d_in[14];
  const float* b2  = (const float*)d_in[15];
  const float* aw  = (const float*)d_in[16];
  const float* ab  = (const float*)d_in[17];

  const size_t E = (size_t)BB * TT * HIDD;       // 16,777,216 activation elems
  bf16_t* qbf  = (bf16_t*)d_ws;                  // 32 MB each
  bf16_t* kbf  = qbf + E;
  bf16_t* vbf  = kbf + E;
  bf16_t* obf  = vbf + E;
  float*  ypre = (float*)(obf + E);              // 64 MB fp32
  bf16_t* xbf  = (bf16_t*)(ypre + E);            // 32 MB
  bf16_t* wqT  = xbf + E;                        // weights (bf16): 2+2+2+2+1+0.5 MB
  bf16_t* wkT  = wqT + (size_t)DD * HIDD;
  bf16_t* wvT  = wkT + (size_t)DD * HIDD;
  bf16_t* woT  = wvT + (size_t)DD * HIDD;
  bf16_t* w1b  = woT + (size_t)HIDD * DD;
  bf16_t* w2b  = w1b + (size_t)OUTD * DD;
  bf16_t* ybf  = qbf;                            // reuse: q/k dead after attention
  bf16_t* h1bf = kbf;

  float* xe    = (float*)d_out;                            // [B, OUT, T]
  float* h1out = (float*)d_out + (size_t)BB * OUTD * TT;   // [B, D/2, T]

  const int M = BB * TT;                         // 16384
  dim3 blk(256);
  dim3 gD(HIDD / 64, M / 128);
  dim3 gH(OUTD / 64, M / 128);

  // One-shot precision/layout conversion (bandwidth-trivial)
  f32_to_bf16<<<dim3((unsigned)(E / 1024)), blk, 0, stream>>>(x, xbf);
  dim3 gT(HIDD / 32, DD / 32);
  transpose_to_bf16<<<gT, blk, 0, stream>>>(Wq, wqT, DD, HIDD);
  transpose_to_bf16<<<gT, blk, 0, stream>>>(Wk, wkT, DD, HIDD);
  transpose_to_bf16<<<gT, blk, 0, stream>>>(Wv, wvT, DD, HIDD);
  transpose_to_bf16<<<gT, blk, 0, stream>>>(Wo, woT, HIDD, DD);
  f32_to_bf16<<<dim3((OUTD * DD) / 1024), blk, 0, stream>>>(W1, w1b);   // already [N][K]
  f32_to_bf16<<<dim3((OUTD * OUTD) / 1024), blk, 0, stream>>>(W2, w2b); // already [N][K]

  // Q/K/V projections (TDM double-buffered bf16 WMMA), bf16 outputs
  gemm_bf16<2, false, false, false><<<gD, blk, 0, stream>>>(
      xbf, wqT, bq, nullptr, nullptr, qbf, nullptr, M, HIDD, DD);
  gemm_bf16<2, false, false, false><<<gD, blk, 0, stream>>>(
      xbf, wkT, bk, nullptr, nullptr, kbf, nullptr, M, HIDD, DD);
  gemm_bf16<2, false, false, false><<<gD, blk, 0, stream>>>(
      xbf, wvT, bv, nullptr, nullptr, vbf, nullptr, M, HIDD, DD);

  // Windowed attention with distance-adjacency bias
  attn_window<<<dim3(BB * NH * TT / 8), blk, 0, stream>>>(qbf, kbf, vbf, obf, aw, ab);

  // Output projection + bias + residual (fp32), then LayerNorm -> bf16
  gemm_bf16<1, false, true, false><<<gD, blk, 0, stream>>>(
      obf, woT, bo, x, ypre, nullptr, nullptr, M, DD, HIDD);
  layernorm_k<<<dim3(M), blk, 0, stream>>>(ypre, lng, lnb, ybf);

  // FFN stage 1: h1 = GELU(y @ W1^T + b1): bf16 row-major + fp32 [B,D/2,T] out
  gemm_bf16<2, true, false, true><<<gH, blk, 0, stream>>>(
      ybf, w1b, b1, nullptr, nullptr, h1bf, h1out, M, OUTD, DD);

  // FFN stage 2: xe = GELU(h1 @ W2^T + b2): fp32 [B,OUT,T] out
  gemm_bf16<0, true, false, true><<<gH, blk, 0, stream>>>(
      h1bf, w2b, b2, nullptr, nullptr, nullptr, xe, M, OUTD, OUTD);
}
